// DPDEModel_3083786519328
// MI455X (gfx1250) — matrix operand order repaired and checked by reading, not compiled, not run on hardware
//
#include <hip/hip_runtime.h>
#include <math.h>

// ---------------- CDNA5 WMMA types ----------------
typedef __bf16 bf16_t;
typedef __attribute__((ext_vector_type(16))) bf16_t v16bf;
typedef __attribute__((ext_vector_type(8)))  float  v8f;
typedef __attribute__((ext_vector_type(8)))  unsigned short us8;
typedef __attribute__((ext_vector_type(4)))  float  f4;

#define TILE_M 16
#define THREADS 256
#define AP 260   // fp32 LDS tile row stride (floats); 1040B row = 65*16 -> 16B-aligned fragments
#define BP 272   // bf16 LDS tile row stride (ushorts); 544B row = 34*16 -> 16B-aligned fragments

__device__ __forceinline__ unsigned short f2bf(float f) {
  __bf16 b = (__bf16)f;                       // hardware cvt (RNE)
  return __builtin_bit_cast(unsigned short, b);
}
__device__ __forceinline__ float bf2f(unsigned short h) {
  return __uint_as_float(((unsigned int)h) << 16);
}

union AFrag { unsigned short us[16]; us8 h2[2]; v16bf v; };

// O[16][256] = A[16][256] x B[256][256], B given as bf16 "column slabs": Bw[n*256+k] = B[k][n].
// One wave handles 32 output columns (two 16x16 tiles), K-loop of 8 chunks of 32.
// A-fragment layout (16-bit 16x32): lanes 0-15 -> K 0..15 at M=lane, lanes 16-31 -> K 16..31 (contiguous, same order as B).
// B-fragment layout (16-bit 32x16): lanes 0-15 -> K 0..15 at N=lane, lanes 16-31 -> K 16..31.
template <bool ABF, bool OBF>
__device__ __forceinline__ void tile_gemm(const void* __restrict__ Aany,
                                          const unsigned short* __restrict__ Bw,
                                          void* __restrict__ Oany,
                                          int lane, int wave) {
  const int r  = lane & 15;
  const int hi = lane >> 4;
  const int n0 = (wave * 2 + 0) * 16 + r;
  const int n1 = (wave * 2 + 1) * 16 + r;
  const float* Af = (const float*)Aany;
  const unsigned short* Ab = (const unsigned short*)Aany;
  v8f acc0 = {};
  v8f acc1 = {};
  for (int kc = 0; kc < 8; ++kc) {
    const int kb = kc * 32 + hi * 16;
    AFrag ua;
    if (ABF) {
      // two contiguous 16B halves: K [kb, kb+8) and [kb+8, kb+16)
      const us8* pa = (const us8*)(Ab + r * BP + kb);
      ua.h2[0] = pa[0];
      ua.h2[1] = pa[1];
    } else {
      const f4* pf = (const f4*)(Af + r * AP + kb);
      f4 a0 = pf[0], a1 = pf[1], a2 = pf[2], a3 = pf[3];  // 16 contiguous floats = 4 f4
#pragma unroll
      for (int j = 0; j < 4; ++j) {
        ua.us[j]      = f2bf(a0[j]);
        ua.us[4 + j]  = f2bf(a1[j]);
        ua.us[8 + j]  = f2bf(a2[j]);
        ua.us[12 + j] = f2bf(a3[j]);
      }
    }
    const int kw = kc * 32 + hi * 16;
    v16bf bv0 = *(const v16bf*)(Bw + n0 * 256 + kw);
    v16bf bv1 = *(const v16bf*)(Bw + n1 * 256 + kw);
    __builtin_prefetch((const void*)(Bw + n0 * 256 + ((kc + 1) & 7) * 32), 0, 1);
    acc0 = __builtin_amdgcn_wmma_f32_16x16x32_bf16(false, ua.v, false, bv0, (short)0, acc0, false, false);
    acc1 = __builtin_amdgcn_wmma_f32_16x16x32_bf16(false, ua.v, false, bv1, (short)0, acc1, false, false);
  }
  // C/D layout: VGPR q -> row q (lanes 0-15) / q+8 (lanes 16-31), col = lane&15 within tile
#pragma unroll
  for (int q = 0; q < 8; ++q) {
    const int m = q + hi * 8;
    if (OBF) {
      ((unsigned short*)Oany)[m * BP + n0] = f2bf(acc0[q]);
      ((unsigned short*)Oany)[m * BP + n1] = f2bf(acc1[q]);
    } else {
      ((float*)Oany)[m * AP + n0] = acc0[q];
      ((float*)Oany)[m * AP + n1] = acc1[q];
    }
  }
}

// ---------------- shared-memory layouts ----------------
struct SmemInt {
  float h[4][TILE_M * AP];            // h0..h3 (fp32, feed forward GEMMs + derivative math)
  unsigned short p[3][TILE_M * BP];   // pre-mask backprop vectors p0..p2 (bf16)
  unsigned short Bb[2][TILE_M * BP];  // backprop / tangent ping-pong (bf16)
  unsigned short T[4][TILE_M * BP];   // tangent dh0..dh3 (bf16)
  float X[TILE_M][8];
  float red[THREADS];
  float vrow[TILE_M];
  float gcol[3][TILE_M];              // grad cols 0..2
  float hcol[2][2][TILE_M];           // [dir][col-1][row] Hessian entries
};

struct SmemIni {
  float a[TILE_M * AP];
  float b[TILE_M * AP];
  float X[TILE_M][8];
  float red[THREADS];
  float vrow[TILE_M];
};

// sum_n buf[m][n]*w[n] for 16 rows; 16 lanes per row, fixed-order reduce (deterministic)
__device__ __forceinline__ void rowdot_f(const float* buf, const float* __restrict__ w,
                                         float* red, float* out16, float addconst, int tid) {
  const int m = tid >> 4, c = tid & 15;
  float part = 0.f;
#pragma unroll
  for (int i = 0; i < 16; ++i) { const int n = c + (i << 4); part += buf[m * AP + n] * w[n]; }
  red[tid] = part;
  __syncthreads();
  if (tid < TILE_M) {
    float s = 0.f;
#pragma unroll
    for (int i = 0; i < 16; ++i) s += red[tid * 16 + i];
    out16[tid] = s + addconst;
  }
  __syncthreads();
}

__device__ __forceinline__ void rowdot_bf(const unsigned short* buf, const float* __restrict__ w,
                                          float* red, float* out16, int tid) {
  const int m = tid >> 4, c = tid & 15;
  float part = 0.f;
#pragma unroll
  for (int i = 0; i < 16; ++i) { const int n = c + (i << 4); part += bf2f(buf[m * BP + n]) * w[n]; }
  red[tid] = part;
  __syncthreads();
  if (tid < TILE_M) {
    float s = 0.f;
#pragma unroll
    for (int i = 0; i < 16; ++i) s += red[tid * 16 + i];
    out16[tid] = s;
  }
  __syncthreads();
}

// ---------------- weight prep: bf16 transposed (fwd) + bf16 row-major (bwd) ----------------
__global__ __launch_bounds__(THREADS) void k_prep(const float* __restrict__ W1,
                                                  const float* __restrict__ W2,
                                                  const float* __restrict__ W3,
                                                  unsigned short* __restrict__ wb) {
  const int idx = blockIdx.x * THREADS + threadIdx.x;   // 0 .. 3*65536-1
  const int mi  = idx >> 16;
  const int off = idx & 65535;
  const int k = off >> 8, n = off & 255;
  const float* W = (mi == 0) ? W1 : (mi == 1) ? W2 : W3;
  const unsigned short b = f2bf(W[off]);
  unsigned short* Wt = wb + mi * 131072;     // Bcol[n][k] = W[k][n]  (forward: B = W)
  unsigned short* Wr = Wt + 65536;           // Bcol[n][k] = W[n][k]  (backward: B = W^T)
  Wt[n * 256 + k] = b;
  Wr[off]         = b;
}

// ---------------- interior: fwd + grad + 2 Hessian columns + PDE residual ----------------
__global__ __launch_bounds__(THREADS) void k_interior(
    const float* __restrict__ Xg,
    const float* __restrict__ W0, const float* __restrict__ b0,
    const float* __restrict__ b1, const float* __restrict__ b2, const float* __restrict__ b3,
    const float* __restrict__ W4, const float* __restrict__ b4,
    const unsigned short* __restrict__ Wt1, const unsigned short* __restrict__ Wr1,
    const unsigned short* __restrict__ Wt2, const unsigned short* __restrict__ Wr2,
    const unsigned short* __restrict__ Wt3, const unsigned short* __restrict__ Wr3,
    float* __restrict__ partials, int Ntot) {
  __shared__ SmemInt sm;
  const int tid  = threadIdx.x;
  const int lane = tid & 31;
  const int wave = tid >> 5;
  const int row0 = blockIdx.x * TILE_M;

  if (tid < TILE_M * 8) {
    const int m = tid >> 3, c = tid & 7;
    sm.X[m][c] = (c < 7) ? Xg[(row0 + m) * 7 + c] : 0.f;
  }
  __syncthreads();

  // ---- forward ----
  for (int idx = tid; idx < TILE_M * 256; idx += THREADS) {   // layer 0 (K=7): VALU
    const int m = idx >> 8, n = idx & 255;
    float z = b0[n];
#pragma unroll
    for (int k = 0; k < 7; ++k) z += sm.X[m][k] * W0[k * 256 + n];
    sm.h[0][m * AP + n] = tanhf(z);
  }
  __syncthreads();

  tile_gemm<false, false>(sm.h[0], Wt1, sm.h[1], lane, wave);
  __syncthreads();
  for (int idx = tid; idx < TILE_M * 256; idx += THREADS) {
    const int m = idx >> 8, n = idx & 255;
    sm.h[1][m * AP + n] = tanhf(sm.h[1][m * AP + n] + b1[n]);
  }
  __syncthreads();
  tile_gemm<false, false>(sm.h[1], Wt2, sm.h[2], lane, wave);
  __syncthreads();
  for (int idx = tid; idx < TILE_M * 256; idx += THREADS) {
    const int m = idx >> 8, n = idx & 255;
    sm.h[2][m * AP + n] = tanhf(sm.h[2][m * AP + n] + b2[n]);
  }
  __syncthreads();
  tile_gemm<false, false>(sm.h[2], Wt3, sm.h[3], lane, wave);
  __syncthreads();
  for (int idx = tid; idx < TILE_M * 256; idx += THREADS) {
    const int m = idx >> 8, n = idx & 255;
    sm.h[3][m * AP + n] = tanhf(sm.h[3][m * AP + n] + b3[n]);
  }
  __syncthreads();

  rowdot_f(sm.h[3], W4, sm.red, sm.vrow, b4[0], tid);   // v(X)

  // ---- backward:  g3 = w4*s3 ; p_l = G_{l+1} @ W_{l+1}^T ; g_l = p_l * s_l ----
  for (int idx = tid; idx < TILE_M * 256; idx += THREADS) {
    const int m = idx >> 8, n = idx & 255;
    const float h3v = sm.h[3][m * AP + n];
    sm.Bb[0][m * BP + n] = f2bf(W4[n] * (1.f - h3v * h3v));
  }
  __syncthreads();
  tile_gemm<true, true>(sm.Bb[0], Wr3, sm.p[2], lane, wave);
  __syncthreads();
  for (int idx = tid; idx < TILE_M * 256; idx += THREADS) {
    const int m = idx >> 8, n = idx & 255;
    const float h2v = sm.h[2][m * AP + n];
    sm.Bb[1][m * BP + n] = f2bf(bf2f(sm.p[2][m * BP + n]) * (1.f - h2v * h2v));
  }
  __syncthreads();
  tile_gemm<true, true>(sm.Bb[1], Wr2, sm.p[1], lane, wave);
  __syncthreads();
  for (int idx = tid; idx < TILE_M * 256; idx += THREADS) {
    const int m = idx >> 8, n = idx & 255;
    const float h1v = sm.h[1][m * AP + n];
    sm.Bb[0][m * BP + n] = f2bf(bf2f(sm.p[1][m * BP + n]) * (1.f - h1v * h1v));
  }
  __syncthreads();
  tile_gemm<true, true>(sm.Bb[0], Wr1, sm.p[0], lane, wave);
  __syncthreads();
  for (int idx = tid; idx < TILE_M * 256; idx += THREADS) {
    const int m = idx >> 8, n = idx & 255;
    const float h0v = sm.h[0][m * AP + n];
    sm.Bb[1][m * BP + n] = f2bf(bf2f(sm.p[0][m * BP + n]) * (1.f - h0v * h0v));
  }
  __syncthreads();
  for (int j = 0; j < 3; ++j)                         // grad_X cols 0..2 = g0 . W0[j,:]
    rowdot_bf(sm.Bb[1], W0 + j * 256, sm.red, sm.gcol[j], tid);

  // ---- two Hessian columns: JVP of the gradient along e1, e2 ----
  for (int d = 0; d < 2; ++d) {
    const int e = d + 1;
    for (int idx = tid; idx < TILE_M * 256; idx += THREADS) {  // dh0 = s0 * W0[e,:]
      const int m = idx >> 8, n = idx & 255;
      const float h0v = sm.h[0][m * AP + n];
      sm.T[0][m * BP + n] = f2bf((1.f - h0v * h0v) * W0[e * 256 + n]);
    }
    __syncthreads();
    tile_gemm<true, true>(sm.T[0], Wt1, sm.T[1], lane, wave);
    __syncthreads();
    for (int idx = tid; idx < TILE_M * 256; idx += THREADS) {
      const int m = idx >> 8, n = idx & 255;
      const float h1v = sm.h[1][m * AP + n];
      sm.T[1][m * BP + n] = f2bf(bf2f(sm.T[1][m * BP + n]) * (1.f - h1v * h1v));
    }
    __syncthreads();
    tile_gemm<true, true>(sm.T[1], Wt2, sm.T[2], lane, wave);
    __syncthreads();
    for (int idx = tid; idx < TILE_M * 256; idx += THREADS) {
      const int m = idx >> 8, n = idx & 255;
      const float h2v = sm.h[2][m * AP + n];
      sm.T[2][m * BP + n] = f2bf(bf2f(sm.T[2][m * BP + n]) * (1.f - h2v * h2v));
    }
    __syncthreads();
    tile_gemm<true, true>(sm.T[2], Wt3, sm.T[3], lane, wave);
    __syncthreads();
    for (int idx = tid; idx < TILE_M * 256; idx += THREADS) {
      const int m = idx >> 8, n = idx & 255;
      const float h3v = sm.h[3][m * AP + n];
      sm.T[3][m * BP + n] = f2bf(bf2f(sm.T[3][m * BP + n]) * (1.f - h3v * h3v));
    }
    __syncthreads();
    // tangent backward: dg3 = w4*ds3, ds_l = -2 h_l dh_l; dg_l = (dG @ W^T)*s_l + p_l*ds_l
    for (int idx = tid; idx < TILE_M * 256; idx += THREADS) {
      const int m = idx >> 8, n = idx & 255;
      const float h3v = sm.h[3][m * AP + n];
      sm.Bb[0][m * BP + n] = f2bf(-2.f * W4[n] * h3v * bf2f(sm.T[3][m * BP + n]));
    }
    __syncthreads();
    tile_gemm<true, true>(sm.Bb[0], Wr3, sm.Bb[1], lane, wave);
    __syncthreads();
    for (int idx = tid; idx < TILE_M * 256; idx += THREADS) {
      const int m = idx >> 8, n = idx & 255;
      const float h2v = sm.h[2][m * AP + n];
      const float dg  = bf2f(sm.Bb[1][m * BP + n]) * (1.f - h2v * h2v)
                      + bf2f(sm.p[2][m * BP + n]) * (-2.f * h2v * bf2f(sm.T[2][m * BP + n]));
      sm.Bb[1][m * BP + n] = f2bf(dg);
    }
    __syncthreads();
    tile_gemm<true, true>(sm.Bb[1], Wr2, sm.Bb[0], lane, wave);
    __syncthreads();
    for (int idx = tid; idx < TILE_M * 256; idx += THREADS) {
      const int m = idx >> 8, n = idx & 255;
      const float h1v = sm.h[1][m * AP + n];
      const float dg  = bf2f(sm.Bb[0][m * BP + n]) * (1.f - h1v * h1v)
                      + bf2f(sm.p[1][m * BP + n]) * (-2.f * h1v * bf2f(sm.T[1][m * BP + n]));
      sm.Bb[0][m * BP + n] = f2bf(dg);
    }
    __syncthreads();
    tile_gemm<true, true>(sm.Bb[0], Wr1, sm.Bb[1], lane, wave);
    __syncthreads();
    for (int idx = tid; idx < TILE_M * 256; idx += THREADS) {
      const int m = idx >> 8, n = idx & 255;
      const float h0v = sm.h[0][m * AP + n];
      const float dg  = bf2f(sm.Bb[1][m * BP + n]) * (1.f - h0v * h0v)
                      + bf2f(sm.p[0][m * BP + n]) * (-2.f * h0v * bf2f(sm.T[0][m * BP + n]));
      sm.Bb[1][m * BP + n] = f2bf(dg);
    }
    __syncthreads();
    for (int jj = 0; jj < 2; ++jj)                 // H[:, jj+1] for this direction
      rowdot_bf(sm.Bb[1], W0 + (jj + 1) * 256, sm.red, sm.hcol[d][jj], tid);
  }

  // ---- PDE residual, squared, tile partial ----
  if (tid < TILE_M) {
    const int m = tid;
    const float XMAXc = logf(1450.0f);
    const float XMINc = 2.0f * logf(100.0f) - XMAXc;
    const float DDXc  = 2.0f / (XMAXc - XMINc);
    const float DDTc  = 2.0f / 5.0f;
    const float rr   = 0.01f + (sm.X[m][3] + 1.f) * 0.5f * (0.09f - 0.01f);
    const float sg1  = 0.1f  + (sm.X[m][4] + 1.f) * 0.5f * (0.9f - 0.1f);
    const float sg2  = 0.1f  + (sm.X[m][5] + 1.f) * 0.5f * (0.9f - 0.1f);
    const float rho  = 0.1f  + (sm.X[m][6] + 1.f) * 0.5f * (0.9f - 0.1f);
    const float v_dt   = DDTc * sm.gcol[0][m];
    const float v_dx1  = DDXc * sm.gcol[1][m];
    const float v_dx2  = DDXc * sm.gcol[2][m];
    const float v_d11  = DDXc * DDXc * sm.hcol[0][0][m];  // H1[:,1]
    const float v_d12  = DDXc * DDXc * sm.hcol[0][1][m];  // H1[:,2]
    const float v_d21  = DDXc * DDXc * sm.hcol[1][0][m];  // H2[:,1]
    const float v_d22  = DDXc * DDXc * sm.hcol[1][1][m];  // H2[:,2]
    const float res = v_dt + rr * sm.vrow[m]
                    - (rr - 0.5f * sg1 * sg1) * v_dx1
                    - (rr - 0.5f * sg2 * sg2) * v_dx2
                    - 0.5f * sg1 * sg1 * v_d11
                    - 0.5f * sg2 * sg2 * v_d22
                    - 0.5f * rho * sg1 * sg2 * v_d12
                    - 0.5f * rho * sg2 * sg1 * v_d21;
    sm.red[m] = res * res;
  }
  __syncthreads();
  if (tid == 0) {
    float s = 0.f;
#pragma unroll
    for (int i = 0; i < TILE_M; ++i) s += sm.red[i];
    partials[blockIdx.x] = s / (float)Ntot;
  }
}

// ---------------- initial condition: forward + payoff ----------------
__global__ __launch_bounds__(THREADS) void k_initial(
    const float* __restrict__ Xg,
    const float* __restrict__ W0, const float* __restrict__ b0,
    const float* __restrict__ b1, const float* __restrict__ b2, const float* __restrict__ b3,
    const float* __restrict__ W4, const float* __restrict__ b4,
    const unsigned short* __restrict__ Wt1, const unsigned short* __restrict__ Wt2,
    const unsigned short* __restrict__ Wt3,
    float* __restrict__ partials, int Ntot) {
  __shared__ SmemIni sm;
  const int tid  = threadIdx.x;
  const int lane = tid & 31;
  const int wave = tid >> 5;
  const int row0 = blockIdx.x * TILE_M;

  if (tid < TILE_M * 8) {
    const int m = tid >> 3, c = tid & 7;
    sm.X[m][c] = (c < 7) ? Xg[(row0 + m) * 7 + c] : 0.f;
  }
  __syncthreads();
  for (int idx = tid; idx < TILE_M * 256; idx += THREADS) {
    const int m = idx >> 8, n = idx & 255;
    float z = b0[n];
#pragma unroll
    for (int k = 0; k < 7; ++k) z += sm.X[m][k] * W0[k * 256 + n];
    sm.a[m * AP + n] = tanhf(z);
  }
  __syncthreads();
  tile_gemm<false, false>(sm.a, Wt1, sm.b, lane, wave);
  __syncthreads();
  for (int idx = tid; idx < TILE_M * 256; idx += THREADS) {
    const int m = idx >> 8, n = idx & 255;
    sm.b[m * AP + n] = tanhf(sm.b[m * AP + n] + b1[n]);
  }
  __syncthreads();
  tile_gemm<false, false>(sm.b, Wt2, sm.a, lane, wave);
  __syncthreads();
  for (int idx = tid; idx < TILE_M * 256; idx += THREADS) {
    const int m = idx >> 8, n = idx & 255;
    sm.a[m * AP + n] = tanhf(sm.a[m * AP + n] + b2[n]);
  }
  __syncthreads();
  tile_gemm<false, false>(sm.a, Wt3, sm.b, lane, wave);
  __syncthreads();
  for (int idx = tid; idx < TILE_M * 256; idx += THREADS) {
    const int m = idx >> 8, n = idx & 255;
    sm.b[m * AP + n] = tanhf(sm.b[m * AP + n] + b3[n]);
  }
  __syncthreads();
  {  // v = h3 . W4 + b4
    const int m = tid >> 4, c = tid & 15;
    float part = 0.f;
#pragma unroll
    for (int i = 0; i < 16; ++i) { const int n = c + (i << 4); part += sm.b[m * AP + n] * W4[n]; }
    sm.red[tid] = part;
    __syncthreads();
    if (tid < TILE_M) {
      float s = 0.f;
#pragma unroll
      for (int i = 0; i < 16; ++i) s += sm.red[tid * 16 + i];
      sm.vrow[tid] = s + b4[0];
    }
    __syncthreads();
  }
  if (tid < TILE_M) {
    const int m = tid;
    const float XMAXc = logf(1450.0f);
    const float XMINc = 2.0f * logf(100.0f) - XMAXc;
    const float x1 = XMINc + (sm.X[m][1] + 1.f) * 0.5f * (XMAXc - XMINc);
    const float x2 = XMINc + (sm.X[m][2] + 1.f) * 0.5f * (XMAXc - XMINc);
    const float pay = fmaxf(0.5f * (expf(x1) + expf(x2)) - 100.0f, 0.0f);
    const float diff = sm.vrow[m] - pay;
    sm.red[m] = diff * diff;
  }
  __syncthreads();
  if (tid == 0) {
    float s = 0.f;
#pragma unroll
    for (int i = 0; i < TILE_M; ++i) s += sm.red[i];
    partials[blockIdx.x] = s / (float)Ntot;
  }
}

// ---------------- deterministic final reduction ----------------
__global__ __launch_bounds__(THREADS) void k_reduce(const float* __restrict__ parts, int n,
                                                    float* __restrict__ out) {
  __shared__ float r[THREADS];
  float s = 0.f;
  for (int i = threadIdx.x; i < n; i += THREADS) s += parts[i];
  r[threadIdx.x] = s;
  __syncthreads();
  for (int w = THREADS / 2; w > 0; w >>= 1) {
    if (threadIdx.x < w) r[threadIdx.x] += r[threadIdx.x + w];
    __syncthreads();
  }
  if (threadIdx.x == 0) out[0] = r[0];
}

// ---------------- host launcher ----------------
extern "C" void kernel_launch(void* const* d_in, const int* in_sizes, int n_in,
                              void* d_out, int out_size, void* d_ws, size_t ws_size,
                              hipStream_t stream) {
  const float* Xint = (const float*)d_in[0];
  const float* Xini = (const float*)d_in[1];
  const float* W0 = (const float*)d_in[2];
  const float* b0 = (const float*)d_in[3];
  const float* W1 = (const float*)d_in[4];
  const float* b1 = (const float*)d_in[5];
  const float* W2 = (const float*)d_in[6];
  const float* b2 = (const float*)d_in[7];
  const float* W3 = (const float*)d_in[8];
  const float* b3 = (const float*)d_in[9];
  const float* W4 = (const float*)d_in[10];
  const float* b4 = (const float*)d_in[11];

  const int Ni = in_sizes[0] / 7;
  const int Nb = in_sizes[1] / 7;
  const int blocksI = Ni / TILE_M;
  const int blocksB = Nb / TILE_M;

  // workspace: [partI | partB | bf16 weights: Wt1 Wr1 Wt2 Wr2 Wt3 Wr3]
  float* partI = (float*)d_ws;
  float* partB = partI + blocksI;
  unsigned short* wb = (unsigned short*)(partB + blocksB);
  const unsigned short* Wt1 = wb;
  const unsigned short* Wr1 = wb + 65536;
  const unsigned short* Wt2 = wb + 131072;
  const unsigned short* Wr2 = wb + 196608;
  const unsigned short* Wt3 = wb + 262144;
  const unsigned short* Wr3 = wb + 327680;

  k_prep<<<(3 * 65536) / THREADS, THREADS, 0, stream>>>(W1, W2, W3, wb);
  k_interior<<<blocksI, THREADS, 0, stream>>>(Xint, W0, b0, b1, b2, b3, W4, b4,
                                              Wt1, Wr1, Wt2, Wr2, Wt3, Wr3, partI, Ni);
  k_initial<<<blocksB, THREADS, 0, stream>>>(Xini, W0, b0, b1, b2, b3, W4, b4,
                                             Wt1, Wt2, Wt3, partB, Nb);
  k_reduce<<<1, THREADS, 0, stream>>>(partI, blocksI + blocksB, (float*)d_out);
}
